// UniRep_mLSTM_27642409517170
// MI455X (gfx1250) — compile-verified, matrix operand order checked by reading, not run on hardware
//
#include <hip/hip_runtime.h>
#include <hip/hip_bf16.h>
#include <stdint.h>

// ---------------- problem constants ----------------
#define H_    1900
#define B_    64
#define L_    512
#define E_    10
#define NOUT  25
#define KPAD  2048          // H padded to 64 k-iters of 32
#define KB    64            // k-iterations (K=32 each) over KPAD
#define CHUNK_KI 4          // k-iters per LDS A-chunk (128 K values)
#define NCHUNK   16
#define ASTRIDE  136        // 128 K elems + 8 pad (272B rows, 16B aligned)
#define NWG   120           // persistent workgroups
#define NT1   120           // phase-1 16-col tiles of m (120*16 = 1920)
#define NT2   480           // phase-2 16-col gate tiles (480*16 = 7680)

// ---------------- workspace layout (bytes) ----------------
#define OFF_BP1 ((size_t)0)
#define SZ_BP1  ((size_t)NT1 * KB * 1024)          // 7,864,320
#define OFF_BP2 (OFF_BP1 + SZ_BP1)
#define SZ_BP2  ((size_t)NT2 * KB * 1024)          // 31,457,280
#define OFF_H   (OFF_BP2 + SZ_BP2)
#define SZ_HB   ((size_t)B_ * KPAD * 2)            // 262,144 (bf16)
#define OFF_M   (OFF_H + SZ_HB)
#define OFF_C   (OFF_M + SZ_HB)
#define SZ_CB   ((size_t)B_ * KPAD * 4)            // 524,288 (f32)
#define OFF_CNT (OFF_C + SZ_CB)
// total ~40.4 MB

// ---------------- CDNA5 async global->LDS path (guarded) ----------------
#if __has_builtin(__builtin_amdgcn_global_load_async_to_lds_b128) && \
    __has_builtin(__builtin_amdgcn_s_wait_asynccnt)
#define HAVE_ASYNC 1
#define WAIT_ASYNC(n) __builtin_amdgcn_s_wait_asynccnt(n)
typedef int v4i_ __attribute__((vector_size(16)));
typedef __attribute__((address_space(1))) v4i_* gv4p_t;  // global int4*
typedef __attribute__((address_space(3))) v4i_* lv4p_t;  // LDS int4*
#define GPTR(p) ((gv4p_t)(const void*)(p))
#define LPTR(p) ((lv4p_t)(void*)(p))
#else
#define HAVE_ASYNC 0
#define WAIT_ASYNC(n)
#endif

typedef __attribute__((ext_vector_type(16))) __bf16 v16bf;
typedef __attribute__((ext_vector_type(8)))  float  v8f;

union FragBF { uint4 u[2]; v16bf v; };

__device__ __forceinline__ unsigned short f2bf(float f) {
  union { float f; unsigned u; } x; x.f = f;
  unsigned u = x.u;
  u += 0x7fffu + ((u >> 16) & 1u);   // round-to-nearest-even
  return (unsigned short)(u >> 16);
}

__device__ __forceinline__ float sigmoidf_(float x) {
  return 1.0f / (1.0f + expf(-x));
}

// Monotonic grid-wide barrier (counter zeroed by init kernel each launch).
__device__ __forceinline__ void grid_barrier(unsigned* cnt, unsigned target) {
  __threadfence();
  __syncthreads();
  if (threadIdx.x == 0) {
    atomicAdd(cnt, 1u);
    volatile unsigned* vc = (volatile unsigned*)cnt;
    while (*vc < target) __builtin_amdgcn_s_sleep(1);
  }
  __syncthreads();
  __threadfence();
}

// Stage one 64x128 bf16 A-chunk (16 KB) into an LDS buffer.
// gbase must already point at column offset ch*128 of the row-major
// [64][KPAD] bf16 state buffer. 8 x B128 per lane -> asynccnt += 8 per wave.
__device__ __forceinline__ void stage_chunk(const unsigned short* gbase,
                                            unsigned short* lbase, int tid) {
  int row = tid >> 1, half = tid & 1;
  const unsigned short* src = gbase + (size_t)row * KPAD + half * 64;
  unsigned short* dst = lbase + row * ASTRIDE + half * 64;
#if HAVE_ASYNC
  #pragma unroll
  for (int q = 0; q < 8; ++q)
    __builtin_amdgcn_global_load_async_to_lds_b128(
        GPTR(src + q * 8), LPTR(dst + q * 8), 0, 0);
#else
  #pragma unroll
  for (int q = 0; q < 8; ++q)
    ((uint4*)dst)[q] = ((const uint4*)src)[q];
#endif
}

// ---------------- init: seed state + output ----------------
__global__ void init_buffers(const float* __restrict__ h0,
                             const float* __restrict__ c0,
                             const float* __restrict__ b_lin,
                             unsigned short* __restrict__ hbuf,
                             unsigned short* __restrict__ mbuf,
                             float* __restrict__ cbuf,
                             unsigned* __restrict__ cnt,
                             float* __restrict__ out) {
  int idx = blockIdx.x * blockDim.x + threadIdx.x;
  if (idx < B_ * L_ * NOUT) out[idx] = b_lin[idx % NOUT];
  if (idx < B_ * KPAD) {
    int row = idx >> 11, col = idx & (KPAD - 1);
    float h = (col < H_) ? h0[row * H_ + col] : 0.f;
    float c = (col < H_) ? c0[row * H_ + col] : 0.f;
    hbuf[idx] = f2bf(h);
    mbuf[idx] = 0;
    cbuf[idx] = c;
  }
  if (idx < 4) cnt[idx] = 0;
}

// ---------------- pack recurrent weights to bf16 WMMA B-tiles ----------------
// B-tile (K=32 x N=16) lane layout mirrors the documented 16-bit operand
// striping: lanes 0-15 -> N=lane, K runs {0..7, 16..23}; lanes 16-31 ->
// N=lane-16, K runs {8..15, 24..31}; 2 bf16 per VGPR. Each lane's 16 values
// stored contiguously (32B) -> 1KB per (n16, kb) block.
__global__ void pack_weights(const float* __restrict__ w_mh,
                             const float* __restrict__ w_hm,
                             unsigned short* __restrict__ bpack) {
  int gid  = blockIdx.x * blockDim.x + threadIdx.x;
  int blk  = gid >> 5;
  int lane = gid & 31;
  if (blk >= (NT1 + NT2) * KB) return;

  int kb; size_t wrow; bool valid_col; const float* W;
  if (blk < NT1 * KB) {                       // w_mh: m-col = n
    int n16 = blk / KB; kb = blk % KB;
    int col = n16 * 16 + (lane & 15);
    valid_col = (col < H_);
    wrow = (size_t)col * H_;
    W = w_mh;
  } else {                                    // w_hm, gate-interleaved columns
    int b2 = blk - NT1 * KB;
    int n16 = b2 / KB; kb = b2 % KB;
    int hblk = n16 >> 2, gate = n16 & 3;      // n16 = hblk*4 + gate
    int hcol = hblk * 16 + (lane & 15);
    valid_col = (hcol < H_);
    wrow = ((size_t)gate * H_ + hcol) * (size_t)H_;
    W = w_hm;
  }

  int k0 = (lane & 16) ? 8 : 0;
  alignas(16) unsigned short outv[16];
  #pragma unroll
  for (int i = 0; i < 16; ++i) {
    int K = kb * 32 + k0 + (i < 8 ? i : i + 8);
    float v = (valid_col && K < H_) ? W[wrow + K] : 0.f;
    outv[i] = f2bf(v);
  }
  unsigned short* dst = bpack + (size_t)blk * 512 + lane * 16;
  ((uint4*)dst)[0] = ((const uint4*)outv)[0];
  ((uint4*)dst)[1] = ((const uint4*)outv)[1];
}

// ---------------- persistent mLSTM scan ----------------
__global__ void __launch_bounds__(128)
mlstm_persistent(const int* __restrict__ xb,
                 const float* __restrict__ embed_table,
                 const float* __restrict__ w_mx,
                 const float* __restrict__ w_ih,
                 const float* __restrict__ bvec,
                 const float* __restrict__ w_lin,
                 const unsigned short* __restrict__ bp1,
                 const unsigned short* __restrict__ bp2,
                 unsigned short* __restrict__ hbuf,
                 unsigned short* __restrict__ mbuf,
                 float* __restrict__ cbuf,
                 unsigned* __restrict__ barcnt,
                 float* __restrict__ out) {
  __shared__ unsigned short ldsA[2][64 * ASTRIDE]; // double-buffered A chunks
  __shared__ float embLds[64][12];                 // emb_t, E=10 (+pad)
  __shared__ float ldsG[4][64][16];                // per-wave partials / gates
  __shared__ float ldsH[64][16];                   // h tile for output proj
  __shared__ float ldsWlin[25][16];
  __shared__ float ldsWmx[16][10];

  const int tid  = threadIdx.x;
  const int wg   = blockIdx.x;
  const int wave = tid >> 5;
  const int lane = tid & 31;
  const int lrow = lane & 15;
  const int hi   = (lane >> 4) & 1;

  // stage per-WG constant tiles
  for (int i = tid; i < 25 * 16; i += 128) {
    int j = i / 16, c = i % 16, hcol = wg * 16 + c;
    ldsWlin[j][c] = (hcol < H_) ? w_lin[(size_t)j * H_ + hcol] : 0.f;
  }
  for (int i = tid; i < 16 * 10; i += 128) {
    int c = i / 10, e = i % 10, col = wg * 16 + c;
    ldsWmx[c][e] = (col < H_) ? w_mx[(size_t)col * E_ + e] : 0.f;
  }
  // per-lane gate-input weights (phase 2): gate = wave, hcol = wg*16+lrow
  float wih_r[E_]; float b_r = 0.f;
  {
    int hcol = wg * 16 + lrow;
    if (hcol < H_) {
      size_t grow = (size_t)(wave * H_ + hcol);
      #pragma unroll
      for (int e = 0; e < E_; ++e) wih_r[e] = w_ih[grow * E_ + e];
      b_r = bvec[grow];
    } else {
      #pragma unroll
      for (int e = 0; e < E_; ++e) wih_r[e] = 0.f;
    }
  }

  unsigned bar_target = 0;

  for (int t = 0; t < L_; ++t) {
    __syncthreads();
    // stage emb_t = table[xb[:,t]] (row 0 forced to zero)
    if (tid < 64) {
      int tok = xb[tid * L_ + t];
      #pragma unroll
      for (int e = 0; e < E_; ++e)
        embLds[tid][e] = (tok == 0) ? 0.f : embed_table[tok * E_ + e];
    }

    // ======== Phase 1: m = mx_t .* (h @ w_mh^T), 16 cols/WG, K split 4 ways
    v8f acc[4];
    #pragma unroll
    for (int r = 0; r < 4; ++r) acc[r] = (v8f){0,0,0,0,0,0,0,0};

    stage_chunk(hbuf, ldsA[0], tid);                       // prime buffer 0
    for (int ch = 0; ch < NCHUNK; ++ch) {
      const unsigned short* cur = ldsA[ch & 1];
      if (ch + 1 < NCHUNK) {                               // overlap next fill
        stage_chunk(hbuf + (ch + 1) * (CHUNK_KI * 32), ldsA[(ch + 1) & 1], tid);
        WAIT_ASYNC(8);                                     // chunk ch resident
      } else {
        WAIT_ASYNC(0);
      }
      __syncthreads();
      {
        int kbg = ch * CHUNK_KI + wave;                    // this wave's k-iter
        FragBF fb;
        const unsigned short* bsrc = bp1 + ((size_t)wg * KB + kbg) * 512 + lane * 16;
        fb.u[0] = ((const uint4*)bsrc)[0];
        fb.u[1] = ((const uint4*)bsrc)[1];
        __builtin_prefetch((const void*)(bsrc + 2 * 512), 0, 3);
        #pragma unroll
        for (int rb = 0; rb < 4; ++rb) {
          FragBF fa;
          const unsigned short* as =
              cur + (rb * 16 + lrow) * ASTRIDE + wave * 32 + (hi ? 8 : 0);
          fa.u[0] = ((const uint4*)as)[0];
          fa.u[1] = ((const uint4*)(as + 16))[0];
          acc[rb] = __builtin_amdgcn_wmma_f32_16x16x32_bf16(
              false, fa.v, false, fb.v, (short)0, acc[rb], false, false);
        }
      }
      __syncthreads();
    }
    // dump partials, reduce 4-way, apply mx, emit m (bf16)
    #pragma unroll
    for (int rb = 0; rb < 4; ++rb)
      #pragma unroll
      for (int j = 0; j < 8; ++j)
        ldsG[wave][rb * 16 + j + 8 * hi][lrow] = acc[rb][j];
    __syncthreads();
    for (int p = 0; p < 8; ++p) {
      int idx = tid + 128 * p;               // 1024 = 64 rows x 16 cols
      int row = idx >> 4, c = idx & 15;
      float s = ldsG[0][row][c] + ldsG[1][row][c] + ldsG[2][row][c] + ldsG[3][row][c];
      float mx = 0.f;
      #pragma unroll
      for (int e = 0; e < E_; ++e) mx += embLds[row][e] * ldsWmx[c][e];
      int col = wg * 16 + c;
      if (col < H_) mbuf[(size_t)row * KPAD + col] = f2bf(mx * s);
    }
    bar_target += NWG; grid_barrier(barcnt, bar_target);

    // ======== Phase 2: gates = gx_t + m @ w_hm^T ; wave = gate, full K
    #pragma unroll
    for (int r = 0; r < 4; ++r) acc[r] = (v8f){0,0,0,0,0,0,0,0};
    const int n16 = wg * 4 + wave;

    stage_chunk(mbuf, ldsA[0], tid);                       // prime buffer 0
    for (int ch = 0; ch < NCHUNK; ++ch) {
      const unsigned short* cur = ldsA[ch & 1];
      if (ch + 1 < NCHUNK) {
        stage_chunk(mbuf + (ch + 1) * (CHUNK_KI * 32), ldsA[(ch + 1) & 1], tid);
        WAIT_ASYNC(8);
      } else {
        WAIT_ASYNC(0);
      }
      __syncthreads();
      #pragma unroll
      for (int kk = 0; kk < CHUNK_KI; ++kk) {
        int kbg = ch * CHUNK_KI + kk;
        FragBF fb;
        const unsigned short* bsrc = bp2 + ((size_t)n16 * KB + kbg) * 512 + lane * 16;
        fb.u[0] = ((const uint4*)bsrc)[0];
        fb.u[1] = ((const uint4*)bsrc)[1];
        __builtin_prefetch((const void*)(bsrc + 512), 0, 3);
        #pragma unroll
        for (int rb = 0; rb < 4; ++rb) {
          FragBF fa;
          const unsigned short* as =
              cur + (rb * 16 + lrow) * ASTRIDE + kk * 32 + (hi ? 8 : 0);
          fa.u[0] = ((const uint4*)as)[0];
          fa.u[1] = ((const uint4*)(as + 16))[0];
          acc[rb] = __builtin_amdgcn_wmma_f32_16x16x32_bf16(
              false, fa.v, false, fb.v, (short)0, acc[rb], false, false);
        }
      }
      __syncthreads();
    }
    // add gx (emb . w_ih + b), stash gate tile
    #pragma unroll
    for (int rb = 0; rb < 4; ++rb)
      #pragma unroll
      for (int j = 0; j < 8; ++j) {
        int row = rb * 16 + j + 8 * hi;
        float gx = b_r;
        #pragma unroll
        for (int e = 0; e < E_; ++e) gx += embLds[row][e] * wih_r[e];
        ldsG[wave][row][lrow] = acc[rb][j] + gx;
      }
    __syncthreads();

    // ======== Phase 3: cell update + h store + output-projection partials
    for (int p = 0; p < 8; ++p) {
      int idx = tid + 128 * p;
      int row = idx >> 4, c = idx & 15;
      int hcol = wg * 16 + c;
      float iv = ldsG[0][row][c], fv = ldsG[1][row][c];
      float gv = ldsG[2][row][c], ov = ldsG[3][row][c];
      float cold = cbuf[(size_t)row * KPAD + hcol];
      float cnew = sigmoidf_(fv) * cold + sigmoidf_(iv) * tanhf(gv);
      float hval = sigmoidf_(ov) * tanhf(cnew);
      ldsH[row][c] = (hcol < H_) ? hval : 0.f;
      if (hcol < H_) {
        cbuf[(size_t)row * KPAD + hcol] = cnew;
        hbuf[(size_t)row * KPAD + hcol] = f2bf(hval);
      }
    }
    __syncthreads();
    for (int p = 0; p < 13; ++p) {
      int idx = tid + 128 * p;
      if (idx < 64 * NOUT) {
        int row = idx / NOUT, j = idx % NOUT;
        float s = 0.f;
        #pragma unroll
        for (int c = 0; c < 16; ++c) s += ldsH[row][c] * ldsWlin[j][c];
        atomicAdd(&out[((size_t)row * L_ + t) * NOUT + j], s);
      }
    }
    bar_target += NWG; grid_barrier(barcnt, bar_target);
  }
}

// ---------------- host launcher ----------------
extern "C" void kernel_launch(void* const* d_in, const int* in_sizes, int n_in,
                              void* d_out, int out_size, void* d_ws, size_t ws_size,
                              hipStream_t stream) {
  const int*   xb    = (const int*)  d_in[0];
  // d_in[1] = xb_lens: unused by the reference output
  const float* h0    = (const float*)d_in[2];
  const float* c0    = (const float*)d_in[3];
  const float* etab  = (const float*)d_in[4];
  const float* w_mx  = (const float*)d_in[5];
  const float* w_mh  = (const float*)d_in[6];
  const float* w_ih  = (const float*)d_in[7];
  const float* w_hm  = (const float*)d_in[8];
  const float* bvec  = (const float*)d_in[9];
  const float* w_lin = (const float*)d_in[10];
  const float* b_lin = (const float*)d_in[11];
  float* out = (float*)d_out;

  char* ws = (char*)d_ws;
  unsigned short* bp1  = (unsigned short*)(ws + OFF_BP1);
  unsigned short* bp2  = (unsigned short*)(ws + OFF_BP2);
  unsigned short* hbuf = (unsigned short*)(ws + OFF_H);
  unsigned short* mbuf = (unsigned short*)(ws + OFF_M);
  float*          cbuf = (float*)(ws + OFF_C);
  unsigned*       cnt  = (unsigned*)(ws + OFF_CNT);

  // 1) seed h/c state, zero m, seed out with b_lin, reset barrier counter
  hipLaunchKernelGGL(init_buffers, dim3(3200), dim3(256), 0, stream,
                     h0, c0, b_lin, hbuf, mbuf, cbuf, cnt, out);
  // 2) repack recurrent weights into bf16 WMMA B-tiles (L2-resident, 36 MB)
  hipLaunchKernelGGL(pack_weights, dim3(4800), dim3(256), 0, stream,
                     w_mh, w_hm, bp1);
  // 3) persistent fused scan
  hipLaunchKernelGGL(mlstm_persistent, dim3(NWG), dim3(128), 0, stream,
                     xb, etab, w_mx, w_ih, bvec, w_lin,
                     bp1, bp2, hbuf, mbuf, cbuf, cnt, out);
}
